// MoELayer_18313740550636
// MI455X (gfx1250) — matrix operand order repaired and checked by reading, not compiled, not run on hardware
//
#include <hip/hip_runtime.h>
#include <math.h>

// ---------------- problem constants ----------------
#define HID   1280      // hidden == intermediate
#define TOK   4096      // B*S
#define NEXP  16        // 2 shared + 14 routed, unified
#define NROUT 14
#define TOPK4 4
#define KT32  40        // 1280 / 32  (K tiles per GEMM)
#define NT16  80        // 1280 / 16  (N tiles)
#define MT16  256       // 4096 / 16  (M tiles)

typedef __attribute__((ext_vector_type(16))) __bf16 v16bf;
typedef __attribute__((ext_vector_type(8)))  float  v8f;

__device__ __forceinline__ unsigned short f2bf(float f) {
  unsigned u = __float_as_uint(f);
  u += 0x7fffu + ((u >> 16) & 1u);     // round-to-nearest-even
  return (unsigned short)(u >> 16);
}
__device__ __forceinline__ unsigned pack2(unsigned short a, unsigned short b) {
  return (unsigned)a | ((unsigned)b << 16);
}
__device__ __forceinline__ v8f wmma_bf16(v16bf a, v16bf b, v8f c) {
  // D = A(16x32 bf16) x B(32x16 bf16) + C(f32)
  return __builtin_amdgcn_wmma_f32_16x16x32_bf16(false, a, false, b, (short)0, c,
                                                 false, false);
}

// ---------------- out = x (residual init) ----------------
__global__ __launch_bounds__(256) void init_out_kernel(const float* __restrict__ x,
                                                       float* __restrict__ out) {
  int i = blockIdx.x * 256 + threadIdx.x;
  if (i < TOK * HID) out[i] = x[i];
}

// ---------------- x fp32 -> bf16 A-fragments ----------------
// A-frag (16x32, ISA 7.12.2): lane<16 row M=lane holds K {0..7,16..23};
// lane>=16 row M=lane-16 holds K {8..15,24..31}; 32B per lane, 1KB per frag.
__global__ __launch_bounds__(256) void convx_kernel(const float* __restrict__ x,
                                                    uint4* __restrict__ Xfr) {
  const int lane = threadIdx.x & 31;
  const int frag = blockIdx.x * 8 + (threadIdx.x >> 5);   // < MT16*KT32
  const int mt = frag / KT32, kt = frag - mt * KT32;
  const int m  = mt * 16 + (lane & 15);
  const int kb = kt * 32 + ((lane & 16) ? 8 : 0);
  const float* r0 = x + (size_t)m * HID + kb;
  unsigned w[8];
#pragma unroll
  for (int j = 0; j < 4; ++j) w[j]     = pack2(f2bf(r0[2*j]),      f2bf(r0[2*j+1]));
#pragma unroll
  for (int j = 0; j < 4; ++j) w[4 + j] = pack2(f2bf(r0[16 + 2*j]), f2bf(r0[16 + 2*j+1]));
  uint4* d = Xfr + ((size_t)frag * 32 + lane) * 2;
  d[0] = make_uint4(w[0], w[1], w[2], w[3]);
  d[1] = make_uint4(w[4], w[5], w[6], w[7]);
}

// ---------------- weights fp32 -> bf16 B-fragments ----------------
// B-frag (32x16): lane<16 col N=lane holds K {0..7,16..23};
// lane>=16 col N=lane-16 holds K {8..15,24..31}. 2 values per VGPR.
__global__ __launch_bounds__(256) void convw_kernel(
    const float* __restrict__ W1s, const float* __restrict__ W2s,
    const float* __restrict__ W1r, const float* __restrict__ W2r,
    uint4* __restrict__ W1fr, uint4* __restrict__ W2fr) {
  const int lane = threadIdx.x & 31;
  const int fid  = blockIdx.x * 8 + (threadIdx.x >> 5);   // < 32 * KT32*NT16
  const int mat  = fid / (KT32 * NT16);
  const int f    = fid - mat * (KT32 * NT16);
  const int e = mat >> 1, isw2 = mat & 1;
  const int kt = f / NT16, nt = f - kt * NT16;
  const float* src;
  if (!isw2) src = (e < 2) ? (W1s + (size_t)e * HID * HID) : (W1r + (size_t)(e - 2) * HID * HID);
  else       src = (e < 2) ? (W2s + (size_t)e * HID * HID) : (W2r + (size_t)(e - 2) * HID * HID);
  const int n  = nt * 16 + (lane & 15);
  const int kb = kt * 32 + ((lane & 16) ? 8 : 0);
  const float* p = src + (size_t)kb * HID + n;
  unsigned w[8];
#pragma unroll
  for (int j = 0; j < 8; ++j)
    w[j] = pack2(f2bf(p[(size_t)(2*j + (j >= 4 ? 8 : 0)) * HID]), f2bf(p[(size_t)(2*j+1 + (j >= 4 ? 8 : 0)) * HID]));
  uint4* base = isw2 ? W2fr : W1fr;
  uint4* d = base + (((size_t)e * (KT32 * NT16) + f) * 32 + lane) * 2;
  d[0] = make_uint4(w[0], w[1], w[2], w[3]);
  d[1] = make_uint4(w[4], w[5], w[6], w[7]);
}

// ---------------- unify biases into [16][1280] ----------------
__global__ __launch_bounds__(256) void bias_kernel(
    const float* __restrict__ b1s, const float* __restrict__ b2s,
    const float* __restrict__ b1r, const float* __restrict__ b2r,
    float* __restrict__ b1u, float* __restrict__ b2u) {
  int i = blockIdx.x * 256 + threadIdx.x;
  if (i >= NEXP * HID) return;
  int e = i / HID, j = i - e * HID;
  b1u[i] = (e < 2) ? b1s[e * HID + j] : b1r[(e - 2) * HID + j];
  b2u[i] = (e < 2) ? b2s[e * HID + j] : b2r[(e - 2) * HID + j];
}

// ---------------- router: softmax + top-4 gates -> [TOK][16] ----------------
__global__ __launch_bounds__(256) void router_kernel(
    const float* __restrict__ x, const float* __restrict__ Wr,
    const float* __restrict__ br, float* __restrict__ gates) {
  const int lane = threadIdx.x & 31;
  const int t = blockIdx.x * 8 + (threadIdx.x >> 5);
  float acc[NROUT];
#pragma unroll
  for (int e = 0; e < NROUT; ++e) acc[e] = 0.f;
  const float* xr = x + (size_t)t * HID;
  for (int k = lane; k < HID; k += 32) {
    float xv = xr[k];
    const float* wr = Wr + (size_t)k * NROUT;
#pragma unroll
    for (int e = 0; e < NROUT; ++e) acc[e] = fmaf(xv, wr[e], acc[e]);
  }
#pragma unroll
  for (int e = 0; e < NROUT; ++e) {
#pragma unroll
    for (int off = 16; off >= 1; off >>= 1)
      acc[e] += __shfl_xor(acc[e], off, 32);
  }
  if (lane == 0) {
    float l[NROUT], m = -1e30f;
#pragma unroll
    for (int e = 0; e < NROUT; ++e) { l[e] = acc[e] + br[e]; m = fmaxf(m, l[e]); }
    float p[NROUT], s = 0.f;
#pragma unroll
    for (int e = 0; e < NROUT; ++e) { p[e] = expf(l[e] - m); s += p[e]; }
    float inv = 1.f / s;
#pragma unroll
    for (int e = 0; e < NROUT; ++e) p[e] *= inv;
    float g[NROUT];
#pragma unroll
    for (int e = 0; e < NROUT; ++e) g[e] = 0.f;
    for (int it = 0; it < TOPK4; ++it) {       // top-4, first-index on ties
      int bi = 0; float bv = -1.f;
#pragma unroll
      for (int e = 0; e < NROUT; ++e) if (p[e] > bv) { bv = p[e]; bi = e; }
      g[bi] = bv; p[bi] = -2.f;
    }
    float* gt = gates + (size_t)t * NEXP;
    gt[0] = 1.f; gt[1] = 1.f;                  // shared experts, gate = 1
#pragma unroll
    for (int e = 0; e < NROUT; ++e) gt[2 + e] = g[e];
  }
}

// ---------------- GEMM1: H = gelu(X @ W1[e] + b1[e]) -> bf16 A-frags ----------------
// grid (10, 32, 4): WG tile 128(M) x 128(N); 8 waves as 4(M) x 2(N); wave 32x64.
__global__ __launch_bounds__(256) void gemm1_kernel(
    const v16bf* __restrict__ Xfr, const v16bf* __restrict__ W1fr,
    const float* __restrict__ b1u, unsigned short* __restrict__ Hfr, int eBase) {
  const int lane  = threadIdx.x & 31;
  const int wave  = threadIdx.x >> 5;
  const int mBase = blockIdx.y * 8 + (wave >> 1) * 2;   // 16-row tiles
  const int nBase = blockIdx.x * 8 + (wave & 1) * 4;    // 16-col tiles
  const int e     = eBase + blockIdx.z;
  const v16bf* B = W1fr + (size_t)e * (KT32 * NT16 * 32);

  const v8f vzero = {};
  v8f c[2][4];
#pragma unroll
  for (int mi = 0; mi < 2; ++mi)
#pragma unroll
    for (int ni = 0; ni < 4; ++ni) c[mi][ni] = vzero;

  for (int kt = 0; kt < KT32; ++kt) {
    v16bf a0 = Xfr[((size_t)(mBase + 0) * KT32 + kt) * 32 + lane];
    v16bf a1 = Xfr[((size_t)(mBase + 1) * KT32 + kt) * 32 + lane];
    v16bf b[4];
#pragma unroll
    for (int ni = 0; ni < 4; ++ni)
      b[ni] = B[((size_t)kt * NT16 + (nBase + ni)) * 32 + lane];
#pragma unroll
    for (int ni = 0; ni < 4; ++ni) {
      c[0][ni] = wmma_bf16(a0, b[ni], c[0][ni]);
      c[1][ni] = wmma_bf16(a1, b[ni], c[1][ni]);
    }
  }

  // epilogue: bias + exact-erf gelu, scatter into A-frag layout for GEMM2
  unsigned short* Hs = Hfr + (size_t)blockIdx.z * ((size_t)MT16 * KT32 * 32 * 16);
  const int mrowOff = (lane & 16) ? 8 : 0;
  const int nlane   = lane & 15;
#pragma unroll
  for (int mi = 0; mi < 2; ++mi) {
    const int mt = mBase + mi;
#pragma unroll
    for (int ni = 0; ni < 4; ++ni) {
      const int col  = (nBase + ni) * 16 + nlane;   // intermediate dim
      const float bs = b1u[(size_t)e * HID + col];
      const int kk   = col & 31;
      const int ktI  = col >> 5;
      const int idx  = (kk & 7) + ((kk & 16) ? 8 : 0);
      const int lhi  = (kk & 8) ? 16 : 0;
      unsigned short* dst = Hs + ((size_t)(mt * KT32 + ktI) * 32) * 16 + idx;
#pragma unroll
      for (int r = 0; r < 8; ++r) {
        float v  = c[mi][ni][r] + bs;
        float gl = 0.5f * v * (1.0f + erff(v * 0.70710678118654752f));
        dst[(size_t)(r + mrowOff + lhi) * 16] = f2bf(gl);
      }
    }
  }
}

// ---------------- GEMM2: out += sum_{e in group} gate[t,e]*(H_e @ W2[e] + b2[e]) --------
// grid (10, 32): loops 4 experts internally -> single non-atomic RMW of out.
__global__ __launch_bounds__(256) void gemm2_kernel(
    const v16bf* __restrict__ Hfr, const v16bf* __restrict__ W2fr,
    const float* __restrict__ b2u, const float* __restrict__ gates,
    float* __restrict__ out, int eBase) {
  const int lane  = threadIdx.x & 31;
  const int wave  = threadIdx.x >> 5;
  const int mBase = blockIdx.y * 8 + (wave >> 1) * 2;
  const int nBase = blockIdx.x * 8 + (wave & 1) * 4;
  const int mrowOff = (lane & 16) ? 8 : 0;
  const int nlane   = lane & 15;

  const v8f vzero = {};
  v8f tot[2][4];
#pragma unroll
  for (int mi = 0; mi < 2; ++mi)
#pragma unroll
    for (int ni = 0; ni < 4; ++ni) tot[mi][ni] = vzero;

  for (int es = 0; es < 4; ++es) {
    const int e = eBase + es;
    const v16bf* A = Hfr + (size_t)es * ((size_t)MT16 * KT32 * 32);
    const v16bf* B = W2fr + (size_t)e * (KT32 * NT16 * 32);
    v8f c[2][4];
#pragma unroll
    for (int mi = 0; mi < 2; ++mi)
#pragma unroll
      for (int ni = 0; ni < 4; ++ni) c[mi][ni] = vzero;

    for (int kt = 0; kt < KT32; ++kt) {
      v16bf a0 = A[((size_t)(mBase + 0) * KT32 + kt) * 32 + lane];
      v16bf a1 = A[((size_t)(mBase + 1) * KT32 + kt) * 32 + lane];
      v16bf b[4];
#pragma unroll
      for (int ni = 0; ni < 4; ++ni)
        b[ni] = B[((size_t)kt * NT16 + (nBase + ni)) * 32 + lane];
#pragma unroll
      for (int ni = 0; ni < 4; ++ni) {
        c[0][ni] = wmma_bf16(a0, b[ni], c[0][ni]);
        c[1][ni] = wmma_bf16(a1, b[ni], c[1][ni]);
      }
    }
    // gated accumulate (bias is per-expert, gate-weighted, matching reference)
#pragma unroll
    for (int mi = 0; mi < 2; ++mi) {
      const int t0 = (mBase + mi) * 16 + mrowOff;
      float bs[4];
#pragma unroll
      for (int ni = 0; ni < 4; ++ni)
        bs[ni] = b2u[(size_t)e * HID + (nBase + ni) * 16 + nlane];
#pragma unroll
      for (int r = 0; r < 8; ++r) {
        const float g = gates[(size_t)(t0 + r) * NEXP + e];
#pragma unroll
        for (int ni = 0; ni < 4; ++ni)
          tot[mi][ni][r] += g * (c[mi][ni][r] + bs[ni]);
      }
    }
  }
  // single read-modify-write of out (stream-serialized across launches)
#pragma unroll
  for (int mi = 0; mi < 2; ++mi) {
    const int t0 = (mBase + mi) * 16 + mrowOff;
#pragma unroll
    for (int r = 0; r < 8; ++r) {
#pragma unroll
      for (int ni = 0; ni < 4; ++ni) {
        const size_t o = (size_t)(t0 + r) * HID + (nBase + ni) * 16 + nlane;
        out[o] += tot[mi][ni][r];
      }
    }
  }
}

// ---------------- host-side launch ----------------
extern "C" void kernel_launch(void* const* d_in, const int* in_sizes, int n_in,
                              void* d_out, int out_size, void* d_ws, size_t ws_size,
                              hipStream_t stream) {
  (void)in_sizes; (void)n_in; (void)out_size; (void)ws_size;
  const float* x   = (const float*)d_in[0];
  const float* W1s = (const float*)d_in[1];
  const float* b1s = (const float*)d_in[2];
  const float* W2s = (const float*)d_in[3];
  const float* b2s = (const float*)d_in[4];
  const float* W1r = (const float*)d_in[5];
  const float* b1r = (const float*)d_in[6];
  const float* W2r = (const float*)d_in[7];
  const float* b2r = (const float*)d_in[8];
  const float* Wr  = (const float*)d_in[9];
  const float* br  = (const float*)d_in[10];
  float* out = (float*)d_out;

  char* ws = (char*)d_ws;
  size_t o = 0;
  void* Xfr  = ws + o; o += (size_t)MT16 * KT32 * 1024;            // 10.5 MB
  void* W1fr = ws + o; o += (size_t)NEXP * KT32 * NT16 * 1024;     // 52.4 MB
  void* W2fr = ws + o; o += (size_t)NEXP * KT32 * NT16 * 1024;     // 52.4 MB
  void* Hfr  = ws + o; o += (size_t)4 * MT16 * KT32 * 1024;        // 41.9 MB
  float* gates = (float*)(ws + o); o += (size_t)TOK * NEXP * 4;
  float* b1u   = (float*)(ws + o); o += (size_t)NEXP * HID * 4;
  float* b2u   = (float*)(ws + o); o += (size_t)NEXP * HID * 4;

  init_out_kernel<<<TOK * HID / 256, 256, 0, stream>>>(x, out);
  convx_kernel<<<MT16 * KT32 / 8, 256, 0, stream>>>(x, (uint4*)Xfr);
  convw_kernel<<<2 * NEXP * KT32 * NT16 / 8, 256, 0, stream>>>(
      W1s, W2s, W1r, W2r, (uint4*)W1fr, (uint4*)W2fr);
  bias_kernel<<<NEXP * HID / 256, 256, 0, stream>>>(b1s, b2s, b1r, b2r, b1u, b2u);
  router_kernel<<<TOK / 8, 256, 0, stream>>>(x, Wr, br, gates);

  for (int g = 0; g < 4; ++g) {
    gemm1_kernel<<<dim3(10, 32, 4), 256, 0, stream>>>(
        (const v16bf*)Xfr, (const v16bf*)W1fr, b1u, (unsigned short*)Hfr, g * 4);
    gemm2_kernel<<<dim3(10, 32, 1), 256, 0, stream>>>(
        (const v16bf*)Hfr, (const v16bf*)W2fr, b2u, gates, out, g * 4);
  }
}
